// SelfAttention_71674414236266
// MI455X (gfx1250) — matrix operand order repaired and checked by reading, not compiled, not run on hardware
//
#include <hip/hip_runtime.h>

typedef __attribute__((ext_vector_type(16))) __bf16 v16bf;
typedef __attribute__((ext_vector_type(8)))  __bf16 v8bf;
typedef __attribute__((ext_vector_type(4)))  __bf16 v4bf;
typedef __attribute__((ext_vector_type(8)))  float  v8f;

// Probe for gfx1250 async global->LDS copy (ASYNCcnt path)
#if defined(__has_builtin)
# if __has_builtin(__builtin_amdgcn_global_load_async_to_lds_b128) && \
     __has_builtin(__builtin_amdgcn_s_wait_asynccnt)
#  define USE_ASYNC_LDS 1
# endif
#endif

// The builtin's params are 16-byte int vectors in global (AS1) / LDS (AS3).
typedef __attribute__((__vector_size__(16))) int vi4;
typedef __attribute__((address_space(1))) vi4 g_vi4;
typedef __attribute__((address_space(3))) vi4 l_vi4;

__device__ __forceinline__ v8f wmma_bf16(v16bf a, v16bf b, v8f c) {
    // (neg_a, A, neg_b, B, c_mod, C, reuse_a, reuse_b)
    return __builtin_amdgcn_wmma_f32_16x16x32_bf16(false, a, false, b, (short)0, c, false, false);
}

__device__ __forceinline__ v16bf make_frag(const __bf16* p0, const __bf16* p1) {
    union { v16bf v; v8bf h[2]; } u;
    u.h[0] = *(const v8bf*)p0;
    u.h[1] = *(const v8bf*)p1;
    return u.v;
}

// A-matrix fragment (16x32 bf16): lane<16 -> row, K 0..15; lane>=16 -> K 16..31
__device__ __forceinline__ v16bf frag_a(const __bf16* row_at_k0, int lane) {
    const __bf16* p0 = row_at_k0 + ((lane >> 4) << 4);
    return make_frag(p0, p0 + 8);
}
// B-matrix fragment (32x16 bf16): lane = column; lane<16 -> K 0..15, lane>=16 -> K 16..31
__device__ __forceinline__ v16bf frag_b(const __bf16* col_at_k0, int lane) {
    const __bf16* p0 = col_at_k0 + ((lane >> 4) << 4);
    return make_frag(p0, p0 + 8);
}

// ---------------------------------------------------------------------------
// Kernel 0: zero the output accumulator (out is atomically accumulated into)
// ---------------------------------------------------------------------------
__global__ void zero_out_kernel(float* out, int n) {
    int i = blockIdx.x * blockDim.x + threadIdx.x;
    if (i < n) out[i] = 0.0f;
}

// ---------------------------------------------------------------------------
// Kernel 1: QKV projection GEMM.  M=32768 (b*s), N=512, K=512.
//   z=0 -> Q (scaled by 1/sqrt(U)), row-major bf16 [32768][512]
//   z=1 -> K, row-major bf16 [32768][512]
//   z=2 -> V, stored transposed per batch: Vt[b][u][s]  bf16
// Block tile 128x128, 8 waves (4 M x 2 N), wave tile 32x64 (8 accumulators).
// ---------------------------------------------------------------------------
__global__ __launch_bounds__(256) void qkv_gemm_kernel(
        const float* __restrict__ x,
        const float* __restrict__ Wq,
        const float* __restrict__ Wk,
        const float* __restrict__ Wv,
        __bf16* __restrict__ Qb,
        __bf16* __restrict__ Kb,
        __bf16* __restrict__ Vtb) {
    __shared__ __bf16 As[128][32];   // 8 KB  (x tile, bf16)
    __shared__ __bf16 Bs[128][32];   // 8 KB  (W tile transposed: Bs[n][k])

    const int tid  = threadIdx.x;
    const int lane = tid & 31;
    const int wave = tid >> 5;
    const int wm   = wave >> 1;      // 0..3  (M direction, 32 rows each)
    const int wn   = wave & 1;       // 0..1  (N direction, 64 cols each)
    const int m0   = blockIdx.y * 128;
    const int n0   = blockIdx.x * 128;
    const int z    = blockIdx.z;
    const float* W = (z == 0) ? Wq : ((z == 1) ? Wk : Wv);

    v8f acc[8] = {};

    for (int kk = 0; kk < 512; kk += 32) {
        // --- stage A tile (x fp32 -> bf16): 128x32 ---
        #pragma unroll
        for (int i = 0; i < 4; ++i) {
            int idx = tid + i * 256;            // 0..1023 float4 chunks
            int row = idx >> 3;
            int c4  = (idx & 7) << 2;
            const float4 f = *(const float4*)(x + (size_t)(m0 + row) * 512 + kk + c4);
            v4bf p = { (__bf16)f.x, (__bf16)f.y, (__bf16)f.z, (__bf16)f.w };
            *(v4bf*)&As[row][c4] = p;
        }
        // --- stage B tile (W fp32 -> bf16, transposed): 32x128 -> Bs[128][32] ---
        #pragma unroll
        for (int i = 0; i < 4; ++i) {
            int idx  = tid + i * 256;           // 0..1023 float4 chunks
            int krow = idx >> 5;
            int nc   = (idx & 31) << 2;
            const float4 f = *(const float4*)(W + (size_t)(kk + krow) * 512 + n0 + nc);
            Bs[nc + 0][krow] = (__bf16)f.x;
            Bs[nc + 1][krow] = (__bf16)f.y;
            Bs[nc + 2][krow] = (__bf16)f.z;
            Bs[nc + 3][krow] = (__bf16)f.w;
        }
        __syncthreads();

        v16bf afr[2], bfr[4];
        #pragma unroll
        for (int ms = 0; ms < 2; ++ms)
            afr[ms] = frag_a(&As[wm * 32 + ms * 16 + (lane & 15)][0], lane);
        #pragma unroll
        for (int ns = 0; ns < 4; ++ns)
            bfr[ns] = frag_b(&Bs[wn * 64 + ns * 16 + (lane & 15)][0], lane);
        #pragma unroll
        for (int ms = 0; ms < 2; ++ms)
            #pragma unroll
            for (int ns = 0; ns < 4; ++ns)
                acc[ms * 4 + ns] = wmma_bf16(afr[ms], bfr[ns], acc[ms * 4 + ns]);
        __syncthreads();
    }

    // ---- epilogue: z-branch hoisted OUT of the store loops ----
    if (z == 2) {
        #pragma unroll
        for (int ms = 0; ms < 2; ++ms) {
            #pragma unroll
            for (int ns = 0; ns < 4; ++ns) {
                #pragma unroll
                for (int i = 0; i < 8; ++i) {
                    int m = m0 + wm * 32 + ms * 16 + i + ((lane >> 4) << 3);
                    int n = n0 + wn * 64 + ns * 16 + (lane & 15);
                    int bb = m >> 11;          // batch
                    int s  = m & 2047;         // sequence pos
                    Vtb[((size_t)bb * 512 + n) * 2048 + s] = (__bf16)acc[ms * 4 + ns][i];
                }
            }
        }
    } else {
        __bf16* dst = (z == 0) ? Qb : Kb;
        const float scale = (z == 0) ? 0.04419417382415922f : 1.0f; // fold 1/sqrt(U) into Q
        #pragma unroll
        for (int ms = 0; ms < 2; ++ms) {
            #pragma unroll
            for (int ns = 0; ns < 4; ++ns) {
                #pragma unroll
                for (int i = 0; i < 8; ++i) {
                    int m = m0 + wm * 32 + ms * 16 + i + ((lane >> 4) << 3);
                    int n = n0 + wn * 64 + ns * 16 + (lane & 15);
                    dst[(size_t)m * 512 + n] = (__bf16)(acc[ms * 4 + ns][i] * scale);
                }
            }
        }
    }
}

// ---------------------------------------------------------------------------
// Kernel 2: flash attention + mean over queries.
// Grid: (S/16, B). Block = 256 threads = 8 waves. Each block: 16 queries.
// Each wave owns a 16x64 slice of the context accumulator (4 f32 16x16 tiles).
// Keys processed in chunks of 256 with online softmax.
// ---------------------------------------------------------------------------
__global__ __launch_bounds__(256) void attention_kernel(
        const __bf16* __restrict__ Qb,
        const __bf16* __restrict__ Kb,
        const __bf16* __restrict__ Vtb,
        float* __restrict__ out) {
    constexpr int CHUNK = 256;

    __shared__ __bf16 Qs[16][512];      // 16 KB  query tile
    __shared__ float  Sc[16][CHUNK];    // 16 KB  raw scores for current chunk
    __shared__ __bf16 Ps[16][CHUNK];    //  8 KB  softmax probabilities (bf16)
    __shared__ float  rowM[16], rowL[16], rowC[16];

    const int tid  = threadIdx.x;
    const int lane = tid & 31;
    const int wave = tid >> 5;
    const int b    = blockIdx.y;
    const int qt   = blockIdx.x;
    const int hi8  = (lane >> 4) << 3;   // 0 or 8

    // ---- load Q tile (bf16, pre-scaled by 1/sqrt(U)) into LDS ----
#if defined(USE_ASYNC_LDS)
    #pragma unroll
    for (int i = 0; i < 4; ++i) {
        int idx = tid + i * 256;         // 1024 x 16B chunks
        int row = idx >> 6;
        int c8  = (idx & 63) << 3;
        __builtin_amdgcn_global_load_async_to_lds_b128(
            (g_vi4*)(Qb + ((size_t)(b * 2048 + qt * 16 + row)) * 512 + c8),
            (l_vi4*)&Qs[row][c8], 0, 0);
    }
    __builtin_amdgcn_s_wait_asynccnt(0);
#else
    #pragma unroll
    for (int i = 0; i < 4; ++i) {
        int idx = tid + i * 256;         // 1024 v8bf chunks
        int row = idx >> 6;
        int c8  = (idx & 63) << 3;
        *(v8bf*)&Qs[row][c8] =
            *(const v8bf*)(Qb + ((size_t)(b * 2048 + qt * 16 + row)) * 512 + c8);
    }
#endif
    if (tid < 16) { rowM[tid] = -3.0e38f; rowL[tid] = 0.0f; }
    __syncthreads();

    v8f acc[4] = {};

    for (int kc = 0; kc < 2048 / CHUNK; ++kc) {
        const int key0 = kc * CHUNK;

        // ---- Phase A: scores; each wave does key tiles (wave) and (wave+8),
        //      sharing one Q fragment per K-step between both WMMAs ----
        {
            const __bf16* krow0 =
                Kb + (size_t)(b * 2048 + key0 + wave * 16 + (lane & 15)) * 512;
            const __bf16* krow1 = krow0 + (size_t)128 * 512;   // +8 key tiles
            v8f sc0 = {}, sc1 = {};
            #pragma unroll
            for (int us = 0; us < 16; ++us) {
                v16bf a = frag_a(&Qs[lane & 15][us * 32], lane);
                sc0 = wmma_bf16(a, frag_b(krow0 + us * 32, lane), sc0);
                sc1 = wmma_bf16(a, frag_b(krow1 + us * 32, lane), sc1);
            }
            #pragma unroll
            for (int i = 0; i < 8; ++i) {
                Sc[i + hi8][wave * 16 + (lane & 15)]       = sc0[i];
                Sc[i + hi8][128 + wave * 16 + (lane & 15)] = sc1[i];
            }
        }
        __syncthreads();

        // ---- Phase B: online softmax stats; 2 rows per wave ----
        #pragma unroll
        for (int rr = 0; rr < 2; ++rr) {
            const int r = wave * 2 + rr;
            float vals[8];
            float mx = -3.0e38f;
            #pragma unroll
            for (int c = 0; c < 8; ++c) {
                vals[c] = Sc[r][lane + c * 32];
                mx = fmaxf(mx, vals[c]);
            }
            #pragma unroll
            for (int off = 16; off >= 1; off >>= 1)
                mx = fmaxf(mx, __shfl_xor(mx, off, 32));
            const float mOld = rowM[r];
            const float mNew = fmaxf(mOld, mx);
            float sum = 0.0f;
            #pragma unroll
            for (int c = 0; c < 8; ++c) {
                float p = __expf(vals[c] - mNew);
                Ps[r][lane + c * 32] = (__bf16)p;
                sum += p;
            }
            #pragma unroll
            for (int off = 16; off >= 1; off >>= 1)
                sum += __shfl_xor(sum, off, 32);
            if (lane == 0) {
                float corr = __expf(mOld - mNew);
                rowC[r] = corr;
                rowL[r] = rowL[r] * corr + sum;
                rowM[r] = mNew;
            }
        }
        __syncthreads();

        // ---- Phase C: rescale accumulators, then P x V; one P fragment
        //      per K-step feeds all 4 column tiles ----
        {
            float cf[8];
            #pragma unroll
            for (int i = 0; i < 8; ++i) cf[i] = rowC[i + hi8];
            #pragma unroll
            for (int j = 0; j < 4; ++j)
                #pragma unroll
                for (int i = 0; i < 8; ++i) acc[j][i] *= cf[i];

            const __bf16* vbase =
                Vtb + ((size_t)b * 512 + wave * 64 + (lane & 15)) * 2048 + key0;
            #pragma unroll
            for (int ks = 0; ks < CHUNK / 32; ++ks) {
                v16bf a = frag_a(&Ps[lane & 15][ks * 32], lane);
                #pragma unroll
                for (int j = 0; j < 4; ++j) {
                    const __bf16* bp =
                        vbase + (size_t)j * 16 * 2048 + ks * 32 + ((lane >> 4) << 4);
                    acc[j] = wmma_bf16(a, make_frag(bp, bp + 8), acc[j]);
                }
            }
        }
        __syncthreads();
    }

    // ---- Final: divide by softmax sum, reduce over 16 queries, mean, atomic ----
    float il[8];
    #pragma unroll
    for (int i = 0; i < 8; ++i) il[i] = 1.0f / rowL[i + hi8];

    #pragma unroll
    for (int j = 0; j < 4; ++j) {
        float s = 0.0f;
        #pragma unroll
        for (int i = 0; i < 8; ++i) s += acc[j][i] * il[i];
        s += __shfl_xor(s, 16, 32);     // combine the two query halves
        if (lane < 16)
            atomicAdd(&out[b * 512 + wave * 64 + j * 16 + lane],
                      s * (1.0f / 2048.0f));
    }
}

// ---------------------------------------------------------------------------
extern "C" void kernel_launch(void* const* d_in, const int* in_sizes, int n_in,
                              void* d_out, int out_size, void* d_ws, size_t ws_size,
                              hipStream_t stream) {
    const float* x  = (const float*)d_in[0];
    const float* Wq = (const float*)d_in[1];
    const float* Wk = (const float*)d_in[2];
    const float* Wv = (const float*)d_in[3];
    float* out = (float*)d_out;

    // workspace layout (bf16 elements): Q | K | Vt, each 32768*512 = 16777216
    __bf16* wsb = (__bf16*)d_ws;
    __bf16* Qb  = wsb;
    __bf16* Kb  = wsb + (size_t)16777216;
    __bf16* Vtb = wsb + (size_t)33554432;

    zero_out_kernel<<<(out_size + 255) / 256, 256, 0, stream>>>(out, out_size);

    dim3 g1(512 / 128, 32768 / 128, 3);    // (N tiles, M tiles, {Q,K,V})
    qkv_gemm_kernel<<<g1, 256, 0, stream>>>(x, Wq, Wk, Wv, Qb, Kb, Vtb);

    dim3 g2(2048 / 16, 16);                // (query tiles, batch)
    attention_kernel<<<g2, 256, 0, stream>>>(Qb, Kb, Vtb, out);
}
